// NGAM_57758720196813
// MI455X (gfx1250) — matrix operand order repaired and checked by reading, not compiled, not run on hardware
//
#include <hip/hip_runtime.h>
#include <hip/hip_bf16.h>
#include <stdint.h>

// ---------------------------------------------------------------------------
// NGAM: per-feature scalar-input MLPs fused as grouped GEMMs, CDNA5 WMMA path.
// B=4096 batch, F=128 features, W=128 width, O=32 outputs, 2 hidden layers.
// f16 WMMA inputs, f32 accumulation. Compute-bound (~39 GFLOP vs ~21 MB), so:
//  - prep kernel pre-converts weights to f16, pre-transposed, into d_ws
//  - main kernel double-buffers weight tiles into LDS with
//    GLOBAL_LOAD_ASYNC_TO_LDS_B128 (ASYNCcnt), overlapping staging with WMMA.
// ---------------------------------------------------------------------------

typedef _Float16 half_t;
typedef __attribute__((ext_vector_type(16))) _Float16 v16h;
typedef __attribute__((ext_vector_type(8)))  _Float16 v8h;
typedef __attribute__((ext_vector_type(8)))  float    v8f;

#define NB 4096
#define NF 128
#define NW 128
#define NO 32
#define NL 2

#define ROWS_PER_BLOCK 128   // 8 waves x 16 rows
#define FG 16                // features per block (grid.y = NF/FG = 8)

#define WHT_N (2ull * NF * NW * NW)   // pre-transposed hidden weights, halves
#define WOT_N ((size_t)NF * NO * NW)  // pre-transposed output weights, halves

// Load an A fragment (16x32 f16, contiguous k order, same as B) from a row-major
// [16][128] f16 tile in LDS. lrow = lane&15 (M row), lhi = lane>>4.
//   lanes 0-15 : v0..v7 = K0..15 contiguous
//   lanes16-31 : v0..v7 = K16..31 contiguous
#define LOADA(dst, apbase, kk) do {                                           \
    const half_t* p_ = (apbase) + (kk) * 32;                                  \
    union { v16h v; v8h h[2]; } u_;                                           \
    u_.h[0] = *(const v8h*)(p_);                                              \
    u_.h[1] = *(const v8h*)(p_ + 8);                                          \
    (dst) = u_.v; } while (0)

// 4 K-steps (K=128) of v_wmma_f32_16x16x32_f16 into accv.
// wtbase points at row (ntile*16 + lrow) of the [n][k] LDS weight tile,
// already offset by lhi*16 (lanes 0-15: K0..15 contiguous; 16-31: K16..31).
#define WMMA_K4(accv, wtbase) do {                                            \
    v16h b0_ = *(const v16h*)((wtbase) + 0 * 32);                             \
    accv = __builtin_amdgcn_wmma_f32_16x16x32_f16(false, a0, false, b0_,      \
                                                  (short)0, accv, false, false);\
    v16h b1_ = *(const v16h*)((wtbase) + 1 * 32);                             \
    accv = __builtin_amdgcn_wmma_f32_16x16x32_f16(false, a1, false, b1_,      \
                                                  (short)0, accv, false, false);\
    v16h b2_ = *(const v16h*)((wtbase) + 2 * 32);                             \
    accv = __builtin_amdgcn_wmma_f32_16x16x32_f16(false, a2, false, b2_,      \
                                                  (short)0, accv, false, false);\
    v16h b3_ = *(const v16h*)((wtbase) + 3 * 32);                             \
    accv = __builtin_amdgcn_wmma_f32_16x16x32_f16(false, a3, false, b3_,      \
                                                  (short)0, accv, false, false);\
} while (0)

// Async copy of nchunks*16 bytes global->LDS, spread over 256 threads.
// LDS offset = low 32 bits of the generic pointer.
__device__ __forceinline__ void stage_async(const half_t* __restrict__ g,
                                            half_t* l, int nchunks, int tid) {
    #pragma unroll
    for (int c = tid; c < nchunks; c += 256) {
        uint32_t lds = (uint32_t)(uintptr_t)(l + (size_t)c * 8);
        uint64_t ga  = (uint64_t)(uintptr_t)(g + (size_t)c * 8);
        asm volatile("global_load_async_to_lds_b128 %0, %1, off"
                     :: "v"(lds), "v"(ga) : "memory");
    }
}

__device__ __forceinline__ void wait_async0() {
    asm volatile("s_wait_asynccnt 0x0" ::: "memory");
}

__global__ void ngam_zero_kernel(float* __restrict__ out, int n) {
    int i = blockIdx.x * blockDim.x + threadIdx.x;
    if (i < n) out[i] = 0.0f;
}

// Convert + pre-transpose weights into f16:
//   wht[l][f][v][w] = (f16) Wh[l][f][w][v]
//   wot[f][o][w]    = (f16) Wo[f][w][o]
__global__ void ngam_prep_kernel(const float* __restrict__ Wh,
                                 const float* __restrict__ Wo,
                                 half_t* __restrict__ wht,
                                 half_t* __restrict__ wot) {
    size_t i = (size_t)blockIdx.x * blockDim.x + threadIdx.x;
    if (i < WHT_N) {
        int w  = (int)(i & (NW - 1));
        int v  = (int)((i >> 7) & (NW - 1));
        size_t lf = i >> 14;                       // l*NF + f
        wht[i] = (half_t)Wh[lf * (NW * NW) + (size_t)w * NW + v];
    } else if (i < WHT_N + WOT_N) {
        size_t j = i - WHT_N;
        int w = (int)(j & (NW - 1));
        int o = (int)((j >> 7) & (NO - 1));
        size_t f = j >> 12;                        // NO*NW = 4096 per feature
        wot[j] = (half_t)Wo[f * (NW * NO) + (size_t)w * NO + o];
    }
}

// ---------------------------------------------------------------------------
// Fast path: f16 pre-transposed weights in d_ws, async double-buffered staging
// ---------------------------------------------------------------------------
__global__ __launch_bounds__(256, 2)
void ngam_fast_kernel(const float*  __restrict__ x,    // [B,F]
                      const float*  __restrict__ W1,   // [F,W]
                      const float*  __restrict__ b1,   // [F,W]
                      const half_t* __restrict__ wht,  // [L,F,W(n),W(k)] f16
                      const float*  __restrict__ bh,   // [L,F,W]
                      const half_t* __restrict__ wot,  // [F,O,W(k)] f16
                      const float*  __restrict__ bo,   // [F,O]
                      float* __restrict__ out)         // [B,O]
{
    __shared__ half_t wbuf[2][NW * NW];     // double-buffered weights, 64 KB
    __shared__ half_t hbuf[8][16 * NW];     // per-wave activations, 32 KB

    const int tid  = threadIdx.x;
    const int wave = tid >> 5;
    const int lane = tid & 31;
    const int lrow = lane & 15;
    const int lhi  = lane >> 4;

    const int row0 = blockIdx.x * ROWS_PER_BLOCK + wave * 16;
    const int f0   = blockIdx.y * FG;

    half_t* hb = &hbuf[wave][0];

    v8f oacc0 = {};
    v8f oacc1 = {};

    // Prologue: stage Wh[0][f0] into buffer 0.
    stage_async(wht + (size_t)(0 * NF + f0) * (NW * NW), wbuf[0], 2048, tid);
    wait_async0();
    __syncthreads();
    int p = 0;

    for (int ff = 0; ff < FG; ++ff) {
        const int f = f0 + ff;

        // ---- First layer (elementwise, per-wave): h0 = relu(x*W1 + b1) ----
        for (int i = lane; i < 16 * NW; i += 32) {
            int m = i >> 7, w = i & (NW - 1);
            float xv = x[(size_t)(row0 + m) * NF + f];
            float v  = fmaf(xv, W1[f * NW + w], b1[f * NW + w]);
            hb[i] = (half_t)fmaxf(v, 0.0f);
        }

        // ---- Hidden layers: compute from wbuf[p], prefetch next stage ----
        #pragma unroll
        for (int l = 0; l < NL; ++l) {
            // prefetch next stage's weights into the other buffer
            if (l == 0)
                stage_async(wht + (size_t)(1 * NF + f) * (NW * NW),
                            wbuf[p ^ 1], 2048, tid);
            else
                stage_async(wot + (size_t)f * (NO * NW),
                            wbuf[p ^ 1], 512, tid);

            const half_t* wcur = wbuf[p];

            // A fragments (reads of hb complete before hb is rewritten below)
            v16h a0, a1, a2, a3;
            {
                const half_t* ap = hb + lrow * NW + lhi * 16;
                LOADA(a0, ap, 0); LOADA(a1, ap, 1);
                LOADA(a2, ap, 2); LOADA(a3, ap, 3);
            }

            const float* bhf = bh + ((size_t)l * NF + f) * NW;
            #pragma unroll
            for (int nt = 0; nt < 8; ++nt) {
                const half_t* wtb = wcur + (nt * 16 + lrow) * NW + lhi * 16;
                v8f acc = {};
                WMMA_K4(acc, wtb);

                int n = nt * 16 + lrow;
                float bias = bhf[n];
                int mb = lhi * 8;
                #pragma unroll
                for (int r = 0; r < 8; ++r) {
                    float vv = fmaxf(acc[r] + bias, 0.0f);
                    hb[(mb + r) * NW + n] = (half_t)vv;
                }
            }

            wait_async0();
            __syncthreads();
            p ^= 1;
        }

        // ---- Output layer from wbuf[p]; prefetch next feature's Wh0 ----
        if (ff + 1 < FG)
            stage_async(wht + (size_t)(0 * NF + (f + 1)) * (NW * NW),
                        wbuf[p ^ 1], 2048, tid);

        {
            const half_t* wcur = wbuf[p];
            v16h a0, a1, a2, a3;
            const half_t* ap = hb + lrow * NW + lhi * 16;
            LOADA(a0, ap, 0); LOADA(a1, ap, 1);
            LOADA(a2, ap, 2); LOADA(a3, ap, 3);

            {
                const half_t* wtb = wcur + (0 * 16 + lrow) * NW + lhi * 16;
                WMMA_K4(oacc0, wtb);
                float bias = bo[f * NO + 0 * 16 + lrow];
                #pragma unroll
                for (int r = 0; r < 8; ++r) oacc0[r] += bias;
            }
            {
                const half_t* wtb = wcur + (1 * 16 + lrow) * NW + lhi * 16;
                WMMA_K4(oacc1, wtb);
                float bias = bo[f * NO + 1 * 16 + lrow];
                #pragma unroll
                for (int r = 0; r < 8; ++r) oacc1[r] += bias;
            }
        }

        wait_async0();
        __syncthreads();
        p ^= 1;
    }

    // ---- Combine feature groups: out[b,o] += oacc ----
    {
        int mb = lhi * 8;
        #pragma unroll
        for (int r = 0; r < 8; ++r) {
            size_t row = (size_t)(row0 + mb + r);
            atomicAdd(out + row * NO + lrow,      oacc0[r]);
            atomicAdd(out + row * NO + 16 + lrow, oacc1[r]);
        }
    }
}

// ---------------------------------------------------------------------------
// Fallback (d_ws too small): in-kernel f32->f16 conversion staging (Round 1)
// ---------------------------------------------------------------------------
__global__ __launch_bounds__(256, 2)
void ngam_fallback_kernel(const float* __restrict__ x,
                          const float* __restrict__ W1,
                          const float* __restrict__ b1,
                          const float* __restrict__ Wh,
                          const float* __restrict__ bh,
                          const float* __restrict__ Wo,
                          const float* __restrict__ bo,
                          float* __restrict__ out)
{
    __shared__ half_t wbuf[NW * NW];
    __shared__ half_t hbuf[8][16 * NW];

    const int tid  = threadIdx.x;
    const int wave = tid >> 5;
    const int lane = tid & 31;
    const int lrow = lane & 15;
    const int lhi  = lane >> 4;

    const int row0 = blockIdx.x * ROWS_PER_BLOCK + wave * 16;
    const int f0   = blockIdx.y * FG;

    half_t* hb = &hbuf[wave][0];

    v8f oacc0 = {};
    v8f oacc1 = {};

    for (int ff = 0; ff < FG; ++ff) {
        const int f = f0 + ff;

        for (int i = lane; i < 16 * NW; i += 32) {
            int m = i >> 7, w = i & (NW - 1);
            float xv = x[(size_t)(row0 + m) * NF + f];
            float v  = fmaf(xv, W1[f * NW + w], b1[f * NW + w]);
            hb[i] = (half_t)fmaxf(v, 0.0f);
        }

        for (int l = 0; l < NL; ++l) {
            __syncthreads();
            {
                const float* Wf = Wh + (((size_t)l * NF + f) * NW * NW);
                for (int i = tid; i < NW * NW; i += 256) {
                    int w = i >> 7, v = i & (NW - 1);
                    wbuf[v * NW + w] = (half_t)Wf[i];
                }
            }
            __syncthreads();

            v16h a0, a1, a2, a3;
            {
                const half_t* ap = hb + lrow * NW + lhi * 16;
                LOADA(a0, ap, 0); LOADA(a1, ap, 1);
                LOADA(a2, ap, 2); LOADA(a3, ap, 3);
            }

            const float* bhf = bh + ((size_t)l * NF + f) * NW;
            #pragma unroll
            for (int nt = 0; nt < 8; ++nt) {
                const half_t* wtb = wbuf + (nt * 16 + lrow) * NW + lhi * 16;
                v8f acc = {};
                WMMA_K4(acc, wtb);

                int n = nt * 16 + lrow;
                float bias = bhf[n];
                int mb = lhi * 8;
                #pragma unroll
                for (int r = 0; r < 8; ++r) {
                    float vv = fmaxf(acc[r] + bias, 0.0f);
                    hb[(mb + r) * NW + n] = (half_t)vv;
                }
            }
        }

        __syncthreads();
        {
            const float* Wof = Wo + (size_t)f * NW * NO;
            for (int i = tid; i < NW * NO; i += 256) {
                int w = i >> 5, o = i & (NO - 1);
                wbuf[o * NW + w] = (half_t)Wof[i];
            }
        }
        __syncthreads();

        {
            v16h a0, a1, a2, a3;
            const half_t* ap = hb + lrow * NW + lhi * 16;
            LOADA(a0, ap, 0); LOADA(a1, ap, 1);
            LOADA(a2, ap, 2); LOADA(a3, ap, 3);

            {
                const half_t* wtb = wbuf + (0 * 16 + lrow) * NW + lhi * 16;
                WMMA_K4(oacc0, wtb);
                float bias = bo[f * NO + 0 * 16 + lrow];
                #pragma unroll
                for (int r = 0; r < 8; ++r) oacc0[r] += bias;
            }
            {
                const half_t* wtb = wbuf + (1 * 16 + lrow) * NW + lhi * 16;
                WMMA_K4(oacc1, wtb);
                float bias = bo[f * NO + 1 * 16 + lrow];
                #pragma unroll
                for (int r = 0; r < 8; ++r) oacc1[r] += bias;
            }
        }
    }

    {
        int mb = lhi * 8;
        #pragma unroll
        for (int r = 0; r < 8; ++r) {
            size_t row = (size_t)(row0 + mb + r);
            atomicAdd(out + row * NO + lrow,      oacc0[r]);
            atomicAdd(out + row * NO + 16 + lrow, oacc1[r]);
        }
    }
}

extern "C" void kernel_launch(void* const* d_in, const int* in_sizes, int n_in,
                              void* d_out, int out_size, void* d_ws, size_t ws_size,
                              hipStream_t stream) {
    (void)in_sizes; (void)n_in;
    const float* x  = (const float*)d_in[0];
    const float* W1 = (const float*)d_in[1];
    const float* b1 = (const float*)d_in[2];
    const float* Wh = (const float*)d_in[3];
    const float* bh = (const float*)d_in[4];
    const float* Wo = (const float*)d_in[5];
    const float* bo = (const float*)d_in[6];
    float* out = (float*)d_out;

    ngam_zero_kernel<<<(out_size + 255) / 256, 256, 0, stream>>>(out, out_size);

    dim3 grid(NB / ROWS_PER_BLOCK, NF / FG);   // (32, 8)
    const size_t ws_need = (WHT_N + WOT_N) * sizeof(half_t);

    if (ws_size >= ws_need) {
        half_t* wht = (half_t*)d_ws;
        half_t* wot = wht + WHT_N;
        size_t total = WHT_N + WOT_N;
        ngam_prep_kernel<<<(unsigned)((total + 255) / 256), 256, 0, stream>>>(
            Wh, Wo, wht, wot);
        ngam_fast_kernel<<<grid, 256, 0, stream>>>(x, W1, b1, wht, bh, wot, bo, out);
    } else {
        ngam_fallback_kernel<<<grid, 256, 0, stream>>>(x, W1, b1, Wh, bh, Wo, bo, out);
    }
}
